// MultiHeadSelfAttention2D_39960375722335
// MI455X (gfx1250) — compile-verified, matrix operand order checked by reading, not run on hardware
//
#include <hip/hip_runtime.h>

#define BN     4
#define CC     256
#define NHW    4096
#define NHEADS 4
#define HDIM   64

typedef __attribute__((ext_vector_type(16))) __bf16         v16bf;
typedef __attribute__((ext_vector_type(16))) unsigned short us16;
typedef __attribute__((ext_vector_type(8)))  unsigned int   ui8;
typedef __attribute__((ext_vector_type(4)))  unsigned int   ui4;
typedef __attribute__((ext_vector_type(8)))  float          v8f;

union BfTile {
  v16bf b;
  us16  u;
  ui8   i;
  ui4   q[2];
};

// f32 -> bf16, round-half-up (bias then take high 16 bits)
__device__ __forceinline__ unsigned short f2bf(float f) {
  return (unsigned short)((__builtin_bit_cast(unsigned int, f) + 0x8000u) >> 16);
}

// pack two f32 -> bf16x2 in one dword: two bias adds + one v_perm_b32
__device__ __forceinline__ unsigned int f2bf_pk(float a, float b) {
  unsigned int ua = __builtin_bit_cast(unsigned int, a) + 0x8000u;
  unsigned int ub = __builtin_bit_cast(unsigned int, b) + 0x8000u;
#if defined(__has_builtin) && __has_builtin(__builtin_amdgcn_perm)
  return __builtin_amdgcn_perm(ub, ua, 0x07060302u);
#else
  return (ua >> 16) | (ub & 0xffff0000u);
#endif
}

__device__ __forceinline__ v8f wmma_bf16(const BfTile& a, const BfTile& bm, v8f c) {
  // 8 args: (neg_a, A, neg_b, B, c_mod, C, reuse_a, reuse_b)
  return __builtin_amdgcn_wmma_f32_16x16x32_bf16(false, a.b, false, bm.b,
                                                 (short)0, c, false, false);
}

// K-dim element pair start for VGPR v with the lane-half term folded into the
// base pointer (16-bit A/B operand layout): {0,2,4,6,16,18,20,22}
__device__ __forceinline__ int kpat0(int v) {
  return ((v >> 2) << 4) + ((v & 3) << 1);
}

// Operand whose K-dim is contiguous at `base_half` (= row ptr + half*8):
// two b128 loads at constant offsets.
__device__ __forceinline__ void load_bf16_operand(BfTile& t, const unsigned short* base_half) {
  t.q[0] = *(const ui4*)(base_half);
  t.q[1] = *(const ui4*)(base_half + 16);
}

// ---------------- async global->LDS copy (16B per lane) --------------------
// builtin signature (from compiler diagnostic): takes <4 x int> vector ptrs,
// global (AS1) source and LDS (AS3) destination, plus two imm ints.
typedef int gcc_i4 __attribute__((vector_size(16)));
typedef __attribute__((address_space(1))) gcc_i4 gbl_i4;
typedef __attribute__((address_space(3))) gcc_i4 lds_i4;

#if defined(__has_builtin) && __has_builtin(__builtin_amdgcn_global_load_async_to_lds_b128)
__device__ __forceinline__ void copy16_async(const void* g, void* l) {
  // generic LDS pointers carry the 32-bit LDS offset in the low word
  __builtin_amdgcn_global_load_async_to_lds_b128(
      (gbl_i4*)(unsigned long long)g,
      (lds_i4*)(unsigned int)(unsigned long long)l, 0, 0);
}
__device__ __forceinline__ void async_wait0() {
#if __has_builtin(__builtin_amdgcn_s_wait_asynccnt)
  __builtin_amdgcn_s_wait_asynccnt(0);
#else
  asm volatile("s_wait_asynccnt 0x0" ::: "memory");
#endif
}
#else
__device__ __forceinline__ void copy16_async(const void* g, void* l) {
  *(ui4*)l = *(const ui4*)g;                  // global b128 -> ds_store_b128
}
__device__ __forceinline__ void async_wait0() {}
#endif

// ---------------------------------------------------------------------------
// Kernel 1: fused QKV projection (1x1 conv == per-pixel GEMM), f32 in, bf16
// out.  Q,K stored transposed [b][head][n][d] (d contiguous); V [b][head][d][m].
// ---------------------------------------------------------------------------
__global__ void __launch_bounds__(128)
qkv_proj_kernel(const float* __restrict__ x,
                const float* __restrict__ wq, const float* __restrict__ bq,
                const float* __restrict__ wk, const float* __restrict__ bk,
                const float* __restrict__ wv, const float* __restrict__ bv,
                unsigned short* __restrict__ qt,
                unsigned short* __restrict__ kt,
                unsigned short* __restrict__ vv) {
  const int lane = threadIdx.x & 31;
  const int wid  = threadIdx.x >> 5;
  const int wg   = blockIdx.x * 4 + wid;         // 0 .. 12287
  const int b    = wg / 3072;
  int rem = wg % 3072;
  const int mat  = rem / 1024;  rem %= 1024;     // 0=Q 1=K 2=V
  const int ot   = rem / 64;                     // 16-row output tile
  const int nt   = rem % 64;                     // 64-pixel tile

  const float* w    = (mat == 0) ? wq : (mat == 1) ? wk : wv;
  const float* bias = (mat == 0) ? bq : (mat == 1) ? bk : bv;

  const int row  = lane & 15;
  const int half = lane >> 4;

  const float* wp2 = w + (ot * 16 + row) * CC + half * 8;
  const float* xp  = x + (unsigned long long)b * CC * NHW + nt * 64 + row
                       + (unsigned long long)(half * 8) * NHW;

  v8f acc[4] = {};

  for (int kk = 0; kk < 8; ++kk) {
    BfTile A;
#pragma unroll
    for (int v = 0; v < 8; ++v) {
      float2 ab = *(const float2*)(wp2 + kpat0(v));
      A.i[v] = f2bf_pk(ab.x, ab.y);
    }
#pragma unroll
    for (int ns = 0; ns < 4; ++ns) {
      BfTile Bm;
#pragma unroll
      for (int v = 0; v < 8; ++v) {
        const int off = kpat0(v) * NHW + ns * 16;
        Bm.i[v] = f2bf_pk(xp[off], xp[off + NHW]);
      }
      acc[ns] = wmma_bf16(A, Bm, acc[ns]);
    }
    wp2 += 32;
    xp  += 32 * NHW;
  }

  float b8[8];
#pragma unroll
  for (int r = 0; r < 8; ++r) b8[r] = bias[ot * 16 + r + 8 * half];

  const int head = (ot * 16) / HDIM;
  const int dlow = (ot * 16) % HDIM;

  if (mat < 2) {
    unsigned short* qk = ((mat == 0) ? qt : kt) +
        ((unsigned long long)(b * NHEADS + head) * NHW + nt * 64 + row) * HDIM +
        dlow + 8 * half;
#pragma unroll
    for (int ns = 0; ns < 4; ++ns)
#pragma unroll
      for (int r = 0; r < 8; r += 2)
        *(unsigned int*)(qk + ns * 16 * HDIM + r) =
            f2bf_pk(acc[ns][r] + b8[r], acc[ns][r + 1] + b8[r + 1]);
  } else {
    unsigned short* vp = vv +
        ((unsigned long long)(b * NHEADS + head) * HDIM + dlow + 8 * half) * NHW +
        nt * 64 + row;
#pragma unroll
    for (int ns = 0; ns < 4; ++ns)
#pragma unroll
      for (int r = 0; r < 8; ++r)
        vp[r * NHW + ns * 16] = f2bf(acc[ns][r] + b8[r]);
  }
}

// ---------------------------------------------------------------------------
// Kernel 2: flash attention with block-shared, double-buffered async K/V
// staging.  Block = 4 waves = 64 queries of one (b,h); per 32-key block the
// block async-copies K (4KB) and V (4KB) tiles to LDS, all waves consume via
// ds_load_b128.  8 WMMAs per wave per key block.
// ---------------------------------------------------------------------------
#define KROW 72   // padded K row (64 data + 8 pad shorts) -> conflict-free b128
#define VROW 40   // padded V row (32 data + 8 pad shorts)

__global__ void __launch_bounds__(128)
flash_attn_kernel(const unsigned short* __restrict__ qt,
                  const unsigned short* __restrict__ kt,
                  const unsigned short* __restrict__ vv,
                  float* __restrict__ ov) {
  __shared__ __align__(16) unsigned short sK[2][32 * KROW];
  __shared__ __align__(16) unsigned short sV[2][64 * VROW];
  __shared__ __align__(16) float          sS[4][16][36];
  __shared__ __align__(16) unsigned int   sP[4][16][16];
  __shared__ __align__(16) float          sAlpha[4][16];
  __shared__ __align__(16) float          sSum[4][16];

  const int tid  = threadIdx.x;
  const int lane = tid & 31;
  const int wid  = tid >> 5;
  const int row  = lane & 15;
  const int half = lane >> 4;

  const int blk = blockIdx.x;                    // 0 .. 1023
  const int b   = blk >> 8;                      // / (NHEADS*64)
  const int hh  = (blk >> 6) & 3;
  const int qg  = blk & 63;                      // 64-query group
  const unsigned long long bh = (unsigned long long)(b * NHEADS + hh);
  const int n0  = qg * 64 + wid * 16;            // this wave's first query

  // Q tile -> two A operands (d = 0..31 and 32..63), loaded once (4x b128).
  BfTile aQ[2];
  {
    const unsigned short* qrow = qt + (bh * NHW + n0 + row) * HDIM + half * 8;
    load_bf16_operand(aQ[0], qrow);
    load_bf16_operand(aQ[1], qrow + 32);
  }

  // per-thread staging addresses: 2 x 16B of K, 2 x 16B of V per key block
  const unsigned short* ksrc = kt + bh * NHW * HDIM + tid * 16;   // contiguous tile
  const unsigned short* vsrc = vv + bh * HDIM * NHW
                                  + (unsigned long long)(tid >> 1) * NHW
                                  + ((2 * tid) & 3) * 8;
  const int kdo = (tid >> 2) * KROW + ((2 * tid) & 7) * 8;
  const int vdo = (tid >> 1) * VROW + ((2 * tid) & 3) * 8;

  // prologue: stage key block 0 into buffer 0
  copy16_async(ksrc,     &sK[0][kdo]);
  copy16_async(ksrc + 8, &sK[0][kdo + 8]);
  copy16_async(vsrc,     &sV[0][vdo]);
  copy16_async(vsrc + 8, &sV[0][vdo + 8]);
  async_wait0();
  __syncthreads();

  v8f O[4] = {};
  float rmax = -1e30f, rsum = 0.f;
  const float scale = 0.125f;                    // 1/sqrt(64)
  int buf = 0;

#pragma unroll 1
  for (int kb = 0; kb < NHW / 32; ++kb) {
    // ---- stage NEXT key block into the other buffer (overlaps compute) ----
    if (kb + 1 < NHW / 32) {
      ksrc += 32 * HDIM;
      vsrc += 32;
      copy16_async(ksrc,     &sK[buf ^ 1][kdo]);
      copy16_async(ksrc + 8, &sK[buf ^ 1][kdo + 8]);
      copy16_async(vsrc,     &sV[buf ^ 1][vdo]);
      copy16_async(vsrc + 8, &sV[buf ^ 1][vdo + 8]);
    }

    const unsigned short* skb = sK[buf];
    const unsigned short* svb = sV[buf];

    // ---- S = (Q^T K) * scale : two 16x16 tiles over a 32-key block --------
    v8f S[2];
#pragma unroll
    for (int t = 0; t < 2; ++t) {
      v8f s = {};
#pragma unroll
      for (int g = 0; g < 2; ++g) {
        BfTile Bk;
        load_bf16_operand(Bk, skb + (t * 16 + row) * KROW + g * 32 + half * 8);
        s = wmma_bf16(aQ[g], Bk, s);
      }
      S[t] = s;
    }

    // ---- D-layout -> LDS (scaled) -----------------------------------------
#pragma unroll
    for (int t = 0; t < 2; ++t)
#pragma unroll
      for (int r = 0; r < 8; ++r)
        sS[wid][r + 8 * half][t * 16 + row] = S[t][r] * scale;
    __builtin_amdgcn_wave_barrier();

    // ---- online softmax: one lane per query row; writes packed bf16 P ----
    if (lane < 16) {
      const float4* srow = (const float4*)sS[wid][lane];
      float4 sv4[8];
      float mx = rmax;
#pragma unroll
      for (int j4 = 0; j4 < 8; ++j4) {
        sv4[j4] = srow[j4];
        mx = fmaxf(mx, fmaxf(fmaxf(sv4[j4].x, sv4[j4].y), fmaxf(sv4[j4].z, sv4[j4].w)));
      }
      const float alpha = __expf(rmax - mx);
      float s = rsum * alpha;
      unsigned int* prow = sP[wid][lane];
#pragma unroll
      for (int j4 = 0; j4 < 8; ++j4) {
        float p0 = __expf(sv4[j4].x - mx);
        float p1 = __expf(sv4[j4].y - mx);
        float p2 = __expf(sv4[j4].z - mx);
        float p3 = __expf(sv4[j4].w - mx);
        s += (p0 + p1) + (p2 + p3);
        prow[2 * j4]     = f2bf_pk(p0, p1);
        prow[2 * j4 + 1] = f2bf_pk(p2, p3);
      }
      rmax = mx; rsum = s;
      sAlpha[wid][lane] = alpha;
    }
    __builtin_amdgcn_wave_barrier();

    // ---- rescale O by per-row alpha ---------------------------------------
    float4 al = *(const float4*)&sAlpha[wid][8 * half];
    float4 ah = *(const float4*)&sAlpha[wid][8 * half + 4];
    float a8[8] = {al.x, al.y, al.z, al.w, ah.x, ah.y, ah.z, ah.w};
#pragma unroll
    for (int t = 0; t < 4; ++t)
#pragma unroll
      for (int r = 0; r < 8; ++r) O[t][r] *= a8[r];

    // ---- P A-operand straight from LDS: two ds_load_b128 ------------------
    BfTile P;
    P.q[0] = *(const ui4*)&sP[wid][row][half * 4];
    P.q[1] = *(const ui4*)&sP[wid][row][8 + half * 4];

    // ---- O += P @ V^T : four 16(d)-tiles ----------------------------------
#pragma unroll
    for (int t = 0; t < 4; ++t) {
      BfTile Bv;
      load_bf16_operand(Bv, svb + (t * 16 + row) * VROW + half * 8);
      O[t] = wmma_bf16(P, Bv, O[t]);
    }

    // ---- pipeline barrier: next-stage done, this buffer fully consumed ----
    async_wait0();
    __syncthreads();
    buf ^= 1;
  }

  if (lane < 16) sSum[wid][lane] = rsum;
  __builtin_amdgcn_wave_barrier();
  float4 sl = *(const float4*)&sSum[wid][8 * half];
  float4 sh = *(const float4*)&sSum[wid][8 * half + 4];
  float inv8[8] = {1.f / sl.x, 1.f / sl.y, 1.f / sl.z, 1.f / sl.w,
                   1.f / sh.x, 1.f / sh.y, 1.f / sh.z, 1.f / sh.w};

  float* obase = ov + bh * HDIM * NHW + n0 + 8 * half;
#pragma unroll
  for (int t = 0; t < 4; ++t) {
    const unsigned long long drow = (unsigned long long)(t * 16 + row) * NHW;
#pragma unroll
    for (int r = 0; r < 8; ++r)                  // n-contiguous -> b128 stores
      obase[drow + r] = O[t][r] * inv8[r];
  }
}

// ---------------------------------------------------------------------------
// Kernel 3: output projection + residual: y = gamma*(Wp @ O + bp) + x
// ---------------------------------------------------------------------------
__global__ void __launch_bounds__(128)
out_proj_kernel(const float* __restrict__ ovin,
                const float* __restrict__ wp, const float* __restrict__ bp,
                const float* __restrict__ x,
                const float* __restrict__ gamma,
                float* __restrict__ out) {
  const int lane = threadIdx.x & 31;
  const int wid  = threadIdx.x >> 5;
  const int wg   = blockIdx.x * 4 + wid;         // 0 .. 4095
  const int b    = wg / 1024;
  int rem = wg % 1024;
  const int ot   = rem / 64;
  const int nt   = rem % 64;
  const int row  = lane & 15;
  const int half = lane >> 4;

  const float* wp2 = wp + (ot * 16 + row) * CC + half * 8;
  const float* op  = ovin + (unsigned long long)b * CC * NHW + nt * 64 + row
                          + (unsigned long long)(half * 8) * NHW;

  v8f acc[4] = {};

  for (int kk = 0; kk < 8; ++kk) {
    BfTile A;
#pragma unroll
    for (int v = 0; v < 8; ++v) {
      float2 ab = *(const float2*)(wp2 + kpat0(v));
      A.i[v] = f2bf_pk(ab.x, ab.y);
    }
#pragma unroll
    for (int ns = 0; ns < 4; ++ns) {
      BfTile Bm;
#pragma unroll
      for (int v = 0; v < 8; ++v) {
        const int off = kpat0(v) * NHW + ns * 16;
        Bm.i[v] = f2bf_pk(op[off], op[off + NHW]);
      }
      acc[ns] = wmma_bf16(A, Bm, acc[ns]);
    }
    wp2 += 32;
    op  += 32 * NHW;
  }

  const float g = gamma[0];
  float b8[8];
#pragma unroll
  for (int r = 0; r < 8; ++r) b8[r] = bp[ot * 16 + r + 8 * half];

  const unsigned long long obase =
      (unsigned long long)b * CC * NHW +
      (unsigned long long)(ot * 16 + 8 * half) * NHW + nt * 64 + row;
  const float* xp = x   + obase;
  float*       yp = out + obase;
#pragma unroll
  for (int ns = 0; ns < 4; ++ns)
#pragma unroll
    for (int r = 0; r < 8; ++r) {
      const int off = r * NHW + ns * 16;
      yp[off] = g * (acc[ns][r] + b8[r]) + xp[off];
    }
}

// ---------------------------------------------------------------------------
extern "C" void kernel_launch(void* const* d_in, const int* in_sizes, int n_in,
                              void* d_out, int out_size, void* d_ws, size_t ws_size,
                              hipStream_t stream) {
  (void)in_sizes; (void)n_in; (void)out_size; (void)ws_size;

  const float* x     = (const float*)d_in[0];
  const float* wq    = (const float*)d_in[1];
  const float* bq    = (const float*)d_in[2];
  const float* wk    = (const float*)d_in[3];
  const float* bk    = (const float*)d_in[4];
  const float* wv    = (const float*)d_in[5];
  const float* bv    = (const float*)d_in[6];
  const float* wp    = (const float*)d_in[7];
  const float* bp    = (const float*)d_in[8];
  const float* gamma = (const float*)d_in[9];

  // workspace: Qt (bf16 8MB) | Kt (bf16 8MB) | V (bf16 8MB) | O (f32 16MB)
  const size_t QKV_ELEMS = (size_t)BN * NHEADS * NHW * HDIM;   // 4,194,304
  unsigned short* qt = (unsigned short*)d_ws;
  unsigned short* kt = qt + QKV_ELEMS;
  unsigned short* vv = kt + QKV_ELEMS;
  float*          ov = (float*)(vv + QKV_ELEMS);

  qkv_proj_kernel<<<3072, 128, 0, stream>>>(x, wq, bq, wk, bk, wv, bv, qt, kt, vv);
  flash_attn_kernel<<<1024, 128, 0, stream>>>(qt, kt, vv, ov);
  out_proj_kernel<<<1024, 128, 0, stream>>>(ov, wp, bp, x, gamma, (float*)d_out);
}
